// WindowAttentionRPB_87033217286286
// MI455X (gfx1250) — matrix operand order repaired and checked by reading, not compiled, not run on hardware
//
#include <hip/hip_runtime.h>
#include <hip/hip_bf16.h>

typedef __attribute__((ext_vector_type(16))) _Float16 v16h;
typedef __attribute__((ext_vector_type(8)))  _Float16 v8h;
typedef __attribute__((ext_vector_type(8)))  float    v8f;

#define H_ 4
#define C_ 128
#define D_ 32
#define L_ 64
#define B_ 4096
#define SCALE_ 0.17677669529663687f   // 1/sqrt(32)

// d_ws layout:
//   [0, 131072) bytes : f16 weights, 4 mats x 128 x 128 (Wq*scale, Wk, Wv, Wo), row-major [co][ci]
//   [131072, 196608)  : f32 dense bias  [H][64][64]
#define WSW_HALVES (4 * C_ * C_)

__global__ void prep_kernel(const float* __restrict__ Wq, const float* __restrict__ Wk,
                            const float* __restrict__ Wv, const float* __restrict__ Wo,
                            const float* __restrict__ bias_table, const int* __restrict__ rel_index,
                            _Float16* __restrict__ wsW, float* __restrict__ wsBias) {
    int idx = blockIdx.x * blockDim.x + threadIdx.x;
    if (idx < WSW_HALVES) {
        int mat = idx >> 14;
        int r   = idx & 16383;
        const float* W = (mat == 0) ? Wq : (mat == 1) ? Wk : (mat == 2) ? Wv : Wo;
        float v = W[r];
        if (mat == 0) v *= SCALE_;
        wsW[idx] = (_Float16)v;
    } else {
        int j = idx - WSW_HALVES;
        if (j < H_ * L_ * L_) {
            int h  = j >> 12;          // head
            int qk = j & 4095;         // q*64+k
            wsBias[j] = bias_table[rel_index[qk] * H_ + h];
        }
    }
}

__device__ __forceinline__ v16h make_v16(v8h lo, v8h hi) {
    union { v16h v; v8h h[2]; } u;
    u.h[0] = lo; u.h[1] = hi;
    return u.v;
}

__device__ __forceinline__ v16h ldA(const _Float16* p) {      // A-frag: 16 contiguous halves
    return make_v16(*(const v8h*)p, *(const v8h*)(p + 8));
}
__device__ __forceinline__ v16h ldB(const _Float16* p) {      // B-frag: 16 contiguous halves
    return make_v16(*(const v8h*)p, *(const v8h*)(p + 8));
}

__launch_bounds__(256)
__global__ void win_attn_kernel(const float* __restrict__ q_in,
                                const _Float16* __restrict__ wsW,
                                const float* __restrict__ wsBias,
                                const float* __restrict__ bq, const float* __restrict__ bk,
                                const float* __restrict__ bv, const float* __restrict__ bo,
                                float* __restrict__ out) {
    __shared__ __attribute__((aligned(32))) _Float16 sX [L_][136];   // X, later reused for O (f16)
    __shared__ __attribute__((aligned(32))) _Float16 sQ [L_][136];
    __shared__ __attribute__((aligned(32))) _Float16 sK [L_][136];
    __shared__ __attribute__((aligned(32))) _Float16 sVt[C_][72];    // V transposed: [c][l]
    __shared__ __attribute__((aligned(32))) _Float16 sP [H_][L_][72];

    const int tid  = threadIdx.x;
    const int wave = tid >> 5;
    const int lane = tid & 31;
    const int l16  = lane & 15;
    const int sel  = lane >> 4;           // 0 | 1 (lane half)
    const int b    = blockIdx.x;

    __builtin_prefetch(wsW + (tid << 5), 0, 0);   // pull weights toward caches

    // ---------------- load X tile, convert fp32 -> f16 into LDS ----------------
    {
        const float4* src = (const float4*)(q_in + (size_t)b * (L_ * C_));
        #pragma unroll
        for (int i = 0; i < 8; ++i) {
            int idx = tid + i * 256;            // 2048 float4 total
            int l = idx >> 5;                   // 32 float4 per row
            int c = (idx & 31) << 2;
            float4 f = src[idx];
            sX[l][c + 0] = (_Float16)f.x;
            sX[l][c + 1] = (_Float16)f.y;
            sX[l][c + 2] = (_Float16)f.z;
            sX[l][c + 3] = (_Float16)f.w;
        }
    }
    __syncthreads();

    // ---------------- Q,K,V projections: [64,128] = X @ W^T + b ----------------
    // 96 tiles (3 mats x 4 mb x 8 nb), 12 per wave, K=128 in 4 steps of 32.
    #pragma unroll
    for (int i = 0; i < 12; ++i) {
        int t   = wave + (i << 3);
        int mat = t >> 5;                      // 0=q 1=k 2=v (wave-uniform)
        int rem = t & 31;
        int m0  = (rem >> 3) << 4;
        int n0  = (rem & 7) << 4;
        const float* bvec = (mat == 0) ? bq : (mat == 1) ? bk : bv;
        float bias = bvec[n0 + l16];
        if (mat == 0) bias *= SCALE_;          // scale folded into q (matches Wq*scale)
        v8f acc = { bias, bias, bias, bias, bias, bias, bias, bias };
        #pragma unroll
        for (int k0 = 0; k0 < C_; k0 += 32) {
            v16h a  = ldA(&sX[m0 + l16][k0 + sel * 16]);
            v16h bm = ldB(wsW + (mat << 14) + (n0 + l16) * C_ + k0 + sel * 16);
            acc = __builtin_amdgcn_wmma_f32_16x16x32_f16(false, a, false, bm, (short)0, acc, false, false);
        }
        if (mat == 2) {                         // V stored transposed
            #pragma unroll
            for (int g = 0; g < 8; ++g)
                sVt[n0 + l16][m0 + g + sel * 8] = (_Float16)acc[g];
        } else {
            _Float16 (*dst)[136] = (mat == 0) ? sQ : sK;
            #pragma unroll
            for (int g = 0; g < 8; ++g)
                dst[m0 + g + sel * 8][n0 + l16] = (_Float16)acc[g];
        }
    }
    __syncthreads();

    // ---------------- attention: 2 waves per head, each owns 2x 16-row blocks ----------------
    const int h = wave >> 1;
    #pragma unroll
    for (int rb = 0; rb < 2; ++rb) {
        const int m0 = ((wave & 1) << 5) + (rb << 4);   // 0/16/32/48

        // S = (scaled Q_h) K_h^T + bias, seeded from dense bias table
        v8f s[4];
        const float* bp = wsBias + (h << 12);
        #pragma unroll
        for (int nt = 0; nt < 4; ++nt) {
            int n0 = nt << 4;
            #pragma unroll
            for (int g = 0; g < 8; ++g)
                s[nt][g] = bp[(m0 + g + sel * 8) * L_ + n0 + l16];
        }
        {
            v16h a = ldA(&sQ[m0 + l16][(h << 5) + sel * 16]);  // K-dim = D = 32: one WMMA/tile
            #pragma unroll
            for (int nt = 0; nt < 4; ++nt) {
                v16h bm = ldB(&sK[(nt << 4) + l16][(h << 5) + sel * 16]);
                s[nt] = __builtin_amdgcn_wmma_f32_16x16x32_f16(false, a, false, bm, (short)0, s[nt], false, false);
            }
        }

        // row softmax: rows live in VGPR g across a 16-lane half; xor-reduce within halves
        #pragma unroll
        for (int g = 0; g < 8; ++g) {
            float mx = fmaxf(fmaxf(s[0][g], s[1][g]), fmaxf(s[2][g], s[3][g]));
            #pragma unroll
            for (int d = 1; d < 16; d <<= 1) mx = fmaxf(mx, __shfl_xor(mx, d, 32));
            float e0 = __expf(s[0][g] - mx), e1 = __expf(s[1][g] - mx);
            float e2 = __expf(s[2][g] - mx), e3 = __expf(s[3][g] - mx);
            float sum = e0 + e1 + e2 + e3;
            #pragma unroll
            for (int d = 1; d < 16; d <<= 1) sum += __shfl_xor(sum, d, 32);
            float inv = __builtin_amdgcn_rcpf(sum);
            s[0][g] = e0 * inv; s[1][g] = e1 * inv; s[2][g] = e2 * inv; s[3][g] = e3 * inv;
        }

        // store P (f16) — read back only by this wave
        #pragma unroll
        for (int nt = 0; nt < 4; ++nt)
            #pragma unroll
            for (int g = 0; g < 8; ++g)
                sP[h][m0 + g + sel * 8][(nt << 4) + l16] = (_Float16)s[nt][g];
        asm volatile("s_wait_dscnt 0" ::: "memory");   // in-wave LDS store->load ordering

        // O_h = P @ V_h : rows m0..m0+15, 32 cols, K=64 in 2 steps
        v8f o0 = {}, o1 = {};
        #pragma unroll
        for (int k0 = 0; k0 < L_; k0 += 32) {
            v16h a  = ldA(&sP[h][m0 + l16][k0 + sel * 16]);
            v16h b0 = ldB(&sVt[(h << 5) + l16][k0 + sel * 16]);
            v16h b1 = ldB(&sVt[(h << 5) + 16 + l16][k0 + sel * 16]);
            o0 = __builtin_amdgcn_wmma_f32_16x16x32_f16(false, a, false, b0, (short)0, o0, false, false);
            o1 = __builtin_amdgcn_wmma_f32_16x16x32_f16(false, a, false, b1, (short)0, o1, false, false);
        }
        // pack O (f16) into sX (X no longer needed), head h occupies cols 32h..32h+31
        #pragma unroll
        for (int g = 0; g < 8; ++g) {
            sX[m0 + g + sel * 8][(h << 5) + l16]      = (_Float16)o0[g];
            sX[m0 + g + sel * 8][(h << 5) + 16 + l16] = (_Float16)o1[g];
        }
    }
    __syncthreads();

    // ---------------- output projection: out = O @ Wo^T + bo ----------------
    {
        const int n0 = wave << 4;                       // wave owns one 16-col block
        const float biasO = bo[n0 + l16];
        const _Float16* wBase = wsW + (3 << 14) + (n0 + l16) * C_;
        float* op = out + (size_t)b * (L_ * C_);
        #pragma unroll
        for (int mb = 0; mb < 4; ++mb) {
            int m0 = mb << 4;
            v8f acc = { biasO, biasO, biasO, biasO, biasO, biasO, biasO, biasO };
            #pragma unroll
            for (int k0 = 0; k0 < C_; k0 += 32) {
                v16h a  = ldA(&sX[m0 + l16][k0 + sel * 16]);
                v16h bm = ldB(wBase + k0 + sel * 16);
                acc = __builtin_amdgcn_wmma_f32_16x16x32_f16(false, a, false, bm, (short)0, acc, false, false);
            }
            #pragma unroll
            for (int g = 0; g < 8; ++g)
                op[(m0 + g + sel * 8) * C_ + n0 + l16] = acc[g];
        }
    }
}

extern "C" void kernel_launch(void* const* d_in, const int* in_sizes, int n_in,
                              void* d_out, int out_size, void* d_ws, size_t ws_size,
                              hipStream_t stream) {
    const float* query      = (const float*)d_in[0];
    const float* Wq         = (const float*)d_in[1];
    const float* bq         = (const float*)d_in[2];
    const float* Wk         = (const float*)d_in[3];
    const float* bk         = (const float*)d_in[4];
    const float* Wv         = (const float*)d_in[5];
    const float* bv         = (const float*)d_in[6];
    const float* Wo         = (const float*)d_in[7];
    const float* bo         = (const float*)d_in[8];
    const float* bias_table = (const float*)d_in[9];
    const int*   rel_index  = (const int*)d_in[10];

    _Float16* wsW    = (_Float16*)d_ws;
    float*    wsBias = (float*)((char*)d_ws + WSW_HALVES * sizeof(_Float16));

    // (65536 weight halves + 16384 bias entries) / 256 = 320 blocks exactly
    prep_kernel<<<320, 256, 0, stream>>>(Wq, Wk, Wv, Wo, bias_table, rel_index, wsW, wsBias);
    win_attn_kernel<<<B_, 256, 0, stream>>>(query, wsW, wsBias, bq, bk, bv, bo, (float*)d_out);
}
